// DGCNN_21337397526629
// MI455X (gfx1250) — matrix operand order repaired and checked by reading, not compiled, not run on hardware
//
#include <hip/hip_runtime.h>
#include <hip/hip_bf16.h>
#include <math.h>

// ---------------------------------------------------------------------------
// CDNA5 (gfx1250) wave32 WMMA types & helpers
// ---------------------------------------------------------------------------
typedef __attribute__((ext_vector_type(16))) __bf16 bf16x16;
typedef __attribute__((ext_vector_type(8)))  __bf16 bf16x8;
typedef __attribute__((ext_vector_type(8)))  float  f32x8;
typedef __attribute__((ext_vector_type(2)))  float  f32x2;

#define DEV __device__ __forceinline__

DEV f32x8 wmma_bf16(bf16x16 a, bf16x16 b, f32x8 c) {
  // v_wmma_f32_16x16x32_bf16, D = A(16x32) * B(32x16) + C
  return __builtin_amdgcn_wmma_f32_16x16x32_bf16(false, a, false, b, (short)0, c,
                                                 false, false);
}
DEV f32x8 wmma_f32k4(f32x2 a, f32x2 b, f32x8 c) {
  // v_wmma_f32_16x16x4_f32, D = A(16x4) * B(4x16) + C  (full f32 precision)
  return __builtin_amdgcn_wmma_f32_16x16x4_f32(false, a, false, b, (short)0, c,
                                               false, false);
}

// A-fragment (16x32 bf16, MxK) from a row-major f32 tile (LDS).
// ISA 7.12.2: lane r=lane&15 owns row M=r; per lane the fragment is two runs
// of 8 consecutive K values: kb+8h+0..7 (regs 0-7) and kb+16+8h+0..7 (8-15).
// Contiguous indexing lets the compiler merge into ds_load_b128.
DEV bf16x16 load_a_f32src(const float* base, int stride, int kb, int lane) {
  int r = lane & 15, h = lane >> 4;
  const float* row = base + r * stride + kb + h * 8;
  bf16x16 a;
#pragma unroll
  for (int t = 0; t < 8; ++t) {
    a[t]     = (__bf16)row[t];
    a[8 + t] = (__bf16)row[16 + t];
  }
  return a;
}

// Same A-fragment from a row-major bf16 activation matrix in global memory:
// two aligned 16-byte vector loads per lane.
DEV bf16x16 load_a_bf16src(const __bf16* base, int stride, int kb, int lane) {
  int r = lane & 15, h = lane >> 4;
  const __bf16* row = base + (size_t)r * stride + kb + h * 8;
  union { bf16x16 v; bf16x8 half[2]; } u;
  u.half[0] = *(const bf16x8*)row;
  u.half[1] = *(const bf16x8*)(row + 16);
  return u.v;
}

// B-fragment from a pre-swizzled (fragment-major) weight buffer: lane's 16
// values are 32 contiguous, 32B-aligned bytes -> two b128 loads.
DEV bf16x16 load_b_packed(const __bf16* PW, int frag, int lane) {
  return *(const bf16x16*)(PW + (((size_t)frag << 5) + lane) * 16);
}

// float atomic-max via uint bit pattern -- valid for non-negative values.
DEV void atomic_max_nonneg(float* addr, float v) {
  atomicMax((unsigned int*)addr, __float_as_uint(v));
}

// ---------------------------------------------------------------------------
// Utility kernels
// ---------------------------------------------------------------------------
__global__ void zero_f32_kernel(float* p, int n) {
  int i = blockIdx.x * 256 + threadIdx.x;
  if (i < n) p[i] = 0.0f;
}

// Pack weight W[Kd][Nd] (f32, row-major) into bf16 fragment-major layout:
// frag = (kb/32)*(Nd/16) + nt ; within a fragment lane l holds 16 contiguous
// bf16 = K values kb + 8*(l>>4) + t (t<8), kb + 16 + 8*(l>>4) + t-8 (t>=8) for column n = nt*16 + (l&15).
__global__ void pack_w_kernel(const float* __restrict__ W, __bf16* __restrict__ PW,
                              int Kd, int Nd) {
  int idx = blockIdx.x * 256 + threadIdx.x;
  if (idx >= Kd * Nd) return;
  int frag = idx >> 9;            // / 512  (32 lanes * 16 values)
  int rem  = idx & 511;
  int lane = rem >> 4, t = rem & 15;
  int ntiles = Nd >> 4;
  int kc = frag / ntiles, nt = frag % ntiles;
  int n = nt * 16 + (lane & 15);
  int k = kc * 32 + ((lane >> 4) << 3) + (t & 7) + ((t >> 3) << 4);
  PW[idx] = (__bf16)W[k * Nd + n];
}

__global__ void sqnorm_kernel(const float* __restrict__ F, float* __restrict__ sq, int n) {
  int r = blockIdx.x * 256 + threadIdx.x;
  if (r >= n) return;
  const float* row = F + r * 64;
  float s = 0.f;
#pragma unroll
  for (int k = 0; k < 64; ++k) s += row[k] * row[k];
  sq[r] = s;
}

// s1 = [x1 | y | z]  (N x 192) written as bf16 into the first 192 cols of s2 (ld 1216)
__global__ void concat_s1_kernel(const float* __restrict__ x1, const float* __restrict__ y,
                                 const float* __restrict__ z, __bf16* __restrict__ s2, int n) {
  int idx = blockIdx.x * 256 + threadIdx.x;
  if (idx >= n * 192) return;
  int r = idx / 192, c = idx % 192;
  float v = (c < 64) ? x1[r * 64 + c] : (c < 128 ? y[r * 64 + c - 64] : z[r * 64 + c - 128]);
  s2[(size_t)r * 1216 + c] = (__bf16)v;
}

// ---------------------------------------------------------------------------
// Stage 1: edge MLP (32->64->64, ReLU) + segment_max over dst
// One wave per 16-edge tile; E=160000 -> 10000 tiles, 8 waves/block.
// ---------------------------------------------------------------------------
__global__ void edge_mlp1_kernel(const float* __restrict__ x, const int* __restrict__ ei,
                                 const __bf16* __restrict__ w1a_p, const float* __restrict__ b1a,
                                 const __bf16* __restrict__ w1b_p, const float* __restrict__ b1b,
                                 float* __restrict__ agg, int E) {
  __shared__ float feat[8][16 * 32];
  __shared__ float h1[8][16 * 64];
  __shared__ int   dsts[8][16];
  int wv = threadIdx.x >> 5, lane = threadIdx.x & 31;
  int tile = blockIdx.x * 8 + wv;
  int e0 = tile * 16;
  int r = lane & 15, h = lane >> 4;

  // Stage edge features [xi, xj-xi] for 16 edges (two lanes per row).
  int e = e0 + r;
  int srcI = ei[e];
  int dstI = ei[E + e];
  if (h == 0) dsts[wv][r] = dstI;
  const float* xs = x + srcI * 16;
  const float* xd = x + dstI * 16;
  float* fr = &feat[wv][r * 32];
  if (h == 0) {
#pragma unroll
    for (int k = 0; k < 16; ++k) fr[k] = xd[k];
  } else {
#pragma unroll
    for (int k = 0; k < 16; ++k) fr[16 + k] = xs[k] - xd[k];
  }
  __syncthreads();

  // GEMM1: (16x32) @ (32x64): one A fragment feeds 4 independent WMMAs.
  {
    bf16x16 a1 = load_a_f32src(&feat[wv][0], 32, 0, lane);
    f32x8 acc[4] = {};
#pragma unroll
    for (int nt = 0; nt < 4; ++nt)
      acc[nt] = wmma_bf16(a1, load_b_packed(w1a_p, nt, lane), acc[nt]);
#pragma unroll
    for (int nt = 0; nt < 4; ++nt) {
      int n = nt * 16 + r;
      float bv = b1a[n];
#pragma unroll
      for (int v = 0; v < 8; ++v)
        h1[wv][(h * 8 + v) * 64 + n] = fmaxf(acc[nt][v] + bv, 0.f);
    }
  }
  __syncthreads();

  // GEMM2: (16x64) @ (64x64): K-outer / N-inner, A loaded once per K chunk.
  {
    f32x8 acc[4] = {};
#pragma unroll
    for (int kb = 0; kb < 64; kb += 32) {
      bf16x16 a = load_a_f32src(&h1[wv][0], 64, kb, lane);
#pragma unroll
      for (int nt = 0; nt < 4; ++nt)
        acc[nt] = wmma_bf16(a, load_b_packed(w1b_p, (kb >> 5) * 4 + nt, lane), acc[nt]);
    }
#pragma unroll
    for (int nt = 0; nt < 4; ++nt) {
      int n = nt * 16 + r;
      float bv = b1b[n];
#pragma unroll
      for (int v = 0; v < 8; ++v) {
        int m = h * 8 + v;
        float val = fmaxf(acc[nt][v] + bv, 0.f);
        atomic_max_nonneg(&agg[dsts[wv][m] * 64 + n], val);
      }
    }
  }
}

// ---------------------------------------------------------------------------
// Fused KNN: Gram tile via v_wmma_f32_16x16x4_f32, top-60 per row in LDS.
// One wave per block, one 16-row tile per block (625 blocks). The N x N
// distance matrix (400 MB) is never written to HBM.
// ---------------------------------------------------------------------------
__global__ void knn_topk_kernel(const float* __restrict__ F, const float* __restrict__ sq,
                                int* __restrict__ nbr, int n) {
  const int KNN = 60;
  __shared__ float dtile[16][16];
  __shared__ float best_d[16][KNN];
  __shared__ int   best_i[16][KNN];
  int lane = threadIdx.x;
  int r0 = blockIdx.x * 16;
  const float INF = __int_as_float(0x7f800000);

  for (int idx = lane; idx < 16 * KNN; idx += 32) {
    best_d[idx / KNN][idx % KNN] = INF;
    best_i[idx / KNN][idx % KNN] = 0;
  }
  __syncthreads();

  int rA = lane & 15, h = lane >> 4;
  int mRow = r0 + rA;

  int ctiles = n >> 4;
  for (int ct = 0; ct < ctiles; ++ct) {
    int c0 = ct * 16;
    // Gram tile G = F[r0:r0+16] @ F[c0:c0+16]^T, K=64 in 16 chunks of 4 (f32 WMMA)
    f32x8 c = {};
#pragma unroll
    for (int kb = 0; kb < 64; kb += 4) {
      f32x2 a, b;
      a.x = F[mRow * 64 + kb + 2 * h];
      a.y = F[mRow * 64 + kb + 2 * h + 1];
      int nc = c0 + rA;
      b.x = F[nc * 64 + kb + 2 * h];
      b.y = F[nc * 64 + kb + 2 * h + 1];
      c = wmma_f32k4(a, b, c);
    }
    float sqn = sq[c0 + rA];
#pragma unroll
    for (int v = 0; v < 8; ++v) {
      int m = h * 8 + v;
      int gr = r0 + m, gc = c0 + rA;
      float d = sq[gr] + sqn - 2.f * c[v];
      if (gr == gc) d = INF;  // mask diagonal
      dtile[m][rA] = d;
    }
    __syncthreads();
    // Serial sorted insertion into per-row top-60 (threshold-gated).
    if (lane < 16) {
      float* bd = best_d[lane];
      int*   bi = best_i[lane];
#pragma unroll 4
      for (int t = 0; t < 16; ++t) {
        float d = dtile[lane][t];
        if (d < bd[KNN - 1]) {
          int p = KNN - 1;
          while (p > 0 && bd[p - 1] > d) {
            bd[p] = bd[p - 1];
            bi[p] = bi[p - 1];
            --p;
          }
          bd[p] = d;
          bi[p] = c0 + t;
        }
      }
    }
    __syncthreads();
  }
  for (int idx = lane; idx < 16 * KNN; idx += 32)
    nbr[(r0 + idx / KNN) * KNN + (idx % KNN)] = best_i[idx / KNN][idx % KNN];
}

// ---------------------------------------------------------------------------
// Stage 2: neighbor MLP (128->64->64, ReLU) + max over 60 neighbors (atomic).
// Edges = N*60 = 600000 -> 37500 tiles, 4 waves/block -> 9375 blocks.
// ---------------------------------------------------------------------------
__global__ void edge_mlp2_kernel(const float* __restrict__ X, const int* __restrict__ nbr,
                                 const __bf16* __restrict__ wa_p, const float* __restrict__ ba,
                                 const __bf16* __restrict__ wb_p, const float* __restrict__ bb,
                                 float* __restrict__ out) {
  __shared__ float feat[4][16 * 128];
  __shared__ float h1[4][16 * 64];
  __shared__ int   rowi[4][16];
  int wv = threadIdx.x >> 5, lane = threadIdx.x & 31;
  int tile = blockIdx.x * 4 + wv;
  int e0 = tile * 16;
  int r = lane & 15, h = lane >> 4;

  int e = e0 + r;
  int i = e / 60;
  int j = nbr[e];
  if (h == 0) rowi[wv][r] = i;
  const float* xi = X + i * 64;
  const float* xj = X + j * 64;
  float* fr = &feat[wv][r * 128];
  if (h == 0) {
#pragma unroll
    for (int k = 0; k < 64; ++k) fr[k] = xi[k];
  } else {
#pragma unroll
    for (int k = 0; k < 64; ++k) fr[64 + k] = xj[k] - xi[k];
  }
  __syncthreads();

  // GEMM1: (16x128)@(128x64), K-outer / N-inner with 4 accumulators.
  {
    f32x8 acc[4] = {};
#pragma unroll
    for (int kb = 0; kb < 128; kb += 32) {
      bf16x16 a = load_a_f32src(&feat[wv][0], 128, kb, lane);
#pragma unroll
      for (int nt = 0; nt < 4; ++nt)
        acc[nt] = wmma_bf16(a, load_b_packed(wa_p, (kb >> 5) * 4 + nt, lane), acc[nt]);
    }
#pragma unroll
    for (int nt = 0; nt < 4; ++nt) {
      int ncol = nt * 16 + r;
      float bv = ba[ncol];
#pragma unroll
      for (int v = 0; v < 8; ++v)
        h1[wv][(h * 8 + v) * 64 + ncol] = fmaxf(acc[nt][v] + bv, 0.f);
    }
  }
  __syncthreads();

  // GEMM2: (16x64)@(64x64) + ReLU + neighbor-max by node.
  {
    f32x8 acc[4] = {};
#pragma unroll
    for (int kb = 0; kb < 64; kb += 32) {
      bf16x16 a = load_a_f32src(&h1[wv][0], 64, kb, lane);
#pragma unroll
      for (int nt = 0; nt < 4; ++nt)
        acc[nt] = wmma_bf16(a, load_b_packed(wb_p, (kb >> 5) * 4 + nt, lane), acc[nt]);
    }
#pragma unroll
    for (int nt = 0; nt < 4; ++nt) {
      int ncol = nt * 16 + r;
      float bv = bb[ncol];
#pragma unroll
      for (int v = 0; v < 8; ++v) {
        int m = h * 8 + v;
        float val = fmaxf(acc[nt][v] + bv, 0.f);
        atomic_max_nonneg(&out[rowi[wv][m] * 64 + ncol], val);
      }
    }
  }
}

// Stage 3: single layer (128->64, ReLU) + neighbor max.
__global__ void edge_mlp3_kernel(const float* __restrict__ X, const int* __restrict__ nbr,
                                 const __bf16* __restrict__ w_p, const float* __restrict__ bias,
                                 float* __restrict__ out) {
  __shared__ float feat[4][16 * 128];
  __shared__ int   rowi[4][16];
  int wv = threadIdx.x >> 5, lane = threadIdx.x & 31;
  int tile = blockIdx.x * 4 + wv;
  int e0 = tile * 16;
  int r = lane & 15, h = lane >> 4;

  int e = e0 + r;
  int i = e / 60;
  int j = nbr[e];
  if (h == 0) rowi[wv][r] = i;
  const float* xi = X + i * 64;
  const float* xj = X + j * 64;
  float* fr = &feat[wv][r * 128];
  if (h == 0) {
#pragma unroll
    for (int k = 0; k < 64; ++k) fr[k] = xi[k];
  } else {
#pragma unroll
    for (int k = 0; k < 64; ++k) fr[64 + k] = xj[k] - xi[k];
  }
  __syncthreads();

  {
    f32x8 acc[4] = {};
#pragma unroll
    for (int kb = 0; kb < 128; kb += 32) {
      bf16x16 a = load_a_f32src(&feat[wv][0], 128, kb, lane);
#pragma unroll
      for (int nt = 0; nt < 4; ++nt)
        acc[nt] = wmma_bf16(a, load_b_packed(w_p, (kb >> 5) * 4 + nt, lane), acc[nt]);
    }
#pragma unroll
    for (int nt = 0; nt < 4; ++nt) {
      int ncol = nt * 16 + r;
      float bv = bias[ncol];
#pragma unroll
      for (int v = 0; v < 8; ++v) {
        int m = h * 8 + v;
        float val = fmaxf(acc[nt][v] + bv, 0.f);
        atomic_max_nonneg(&out[rowi[wv][m] * 64 + ncol], val);
      }
    }
  }
}

// ---------------------------------------------------------------------------
// Dense GEMM + bias + optional ReLU, 16x64 output tile per wave:
// out(bf16) = act(A(bf16) @ W(bf16) + b).  One A fragment feeds 4 WMMAs/chunk.
// ---------------------------------------------------------------------------
__global__ void gemm_bias_act_kernel(const __bf16* __restrict__ A, int lda,
                                     const __bf16* __restrict__ PW,
                                     const float* __restrict__ bias,
                                     __bf16* __restrict__ out, int ldo,
                                     int rows, int Kd, int Nd, int do_relu) {
  int wv = threadIdx.x >> 5, lane = threadIdx.x & 31;
  int ctiles = Nd >> 6;  // 64-wide column supertiles
  int gw = blockIdx.x * (blockDim.x >> 5) + wv;
  if (gw >= (rows >> 4) * ctiles) return;
  int rt = gw / ctiles, nt0 = (gw % ctiles) * 4;
  const __bf16* Abase = A + (size_t)(rt * 16) * lda;
  int ntiles = Nd >> 4;

  f32x8 acc[4] = {};
  for (int kb = 0; kb < Kd; kb += 32) {
    if (kb + 32 < Kd)  // gfx1250 global_prefetch of next A chunk
      __builtin_prefetch(Abase + (size_t)(lane & 15) * lda + kb + 32, 0, 3);
    bf16x16 a = load_a_bf16src(Abase, lda, kb, lane);
    int fb = (kb >> 5) * ntiles + nt0;
#pragma unroll
    for (int q = 0; q < 4; ++q)
      acc[q] = wmma_bf16(a, load_b_packed(PW, fb + q, lane), acc[q]);
  }
  int h = lane >> 4, rl = lane & 15;
#pragma unroll
  for (int q = 0; q < 4; ++q) {
    int n = (nt0 + q) * 16 + rl;
    float bv = bias[n];
#pragma unroll
    for (int v = 0; v < 8; ++v) {
      int m = rt * 16 + h * 8 + v;
      float val = acc[q][v] + bv;
      if (do_relu) val = fmaxf(val, 0.f);
      out[(size_t)m * ldo + n] = (__bf16)val;
    }
  }
}

// Final 128->1 layer + sigmoid + BCE loss reduction. out[0]=loss, out[1..N]=b.
__global__ void final_loss_kernel(const __bf16* __restrict__ b3buf,
                                  const float* __restrict__ wf5, const float* __restrict__ bf5,
                                  const float* __restrict__ labels,
                                  float* __restrict__ out, int n) {
  __shared__ float red[256];
  int r = blockIdx.x * 256 + threadIdx.x;
  float contrib = 0.f;
  if (r < n) {
    float acc = bf5[0];
    const __bf16* row = b3buf + (size_t)r * 128;
#pragma unroll
    for (int k = 0; k < 128; ++k) acc += (float)row[k] * wf5[k];
    float p = 1.f / (1.f + __expf(-acc));
    out[1 + r] = p;
    float pc = fminf(fmaxf(p, 1e-7f), 1.f - 1e-7f);
    float lab = labels[r];
    contrib = -(lab * __logf(pc) + (1.f - lab) * __logf(1.f - pc)) / (float)n;
  }
  red[threadIdx.x] = contrib;
  __syncthreads();
  for (int s = 128; s > 0; s >>= 1) {
    if (threadIdx.x < s) red[threadIdx.x] += red[threadIdx.x + s];
    __syncthreads();
  }
  if (threadIdx.x == 0) atomicAdd(&out[0], red[0]);
}

// ---------------------------------------------------------------------------
// Host orchestration
// ---------------------------------------------------------------------------
extern "C" void kernel_launch(void* const* d_in, const int* in_sizes, int n_in,
                              void* d_out, int out_size, void* d_ws, size_t ws_size,
                              hipStream_t stream) {
  const int N = 10000, E = 160000, KNN = 60;

  const float* x      = (const float*)d_in[0];
  const int*   ei     = (const int*)  d_in[1];
  const float* labels = (const float*)d_in[2];
  const float* w1a = (const float*)d_in[3];   const float* b1a = (const float*)d_in[4];
  const float* w1b = (const float*)d_in[5];   const float* b1b = (const float*)d_in[6];
  const float* w2a = (const float*)d_in[7];   const float* b2a = (const float*)d_in[8];
  const float* w2b = (const float*)d_in[9];   const float* b2b = (const float*)d_in[10];
  const float* w3  = (const float*)d_in[11];  const float* b3w = (const float*)d_in[12];
  const float* wf1 = (const float*)d_in[13];  const float* bf1 = (const float*)d_in[14];
  const float* wf2 = (const float*)d_in[15];  const float* bf2 = (const float*)d_in[16];
  const float* wf3 = (const float*)d_in[17];  const float* bf3 = (const float*)d_in[18];
  const float* wf4 = (const float*)d_in[19];  const float* bf4 = (const float*)d_in[20];
  const float* wf5 = (const float*)d_in[21];  const float* bf5 = (const float*)d_in[22];
  float* out = (float*)d_out;

  // Workspace bump allocator (256 B aligned).
  char* p = (char*)d_ws;
  auto alloc = [&](size_t bytes) {
    void* r = (void*)p;
    p += (bytes + 255) & ~(size_t)255;
    return r;
  };
  float*  x1  = (float*)alloc((size_t)N * 64 * 4);
  float*  yb  = (float*)alloc((size_t)N * 64 * 4);
  float*  zb  = (float*)alloc((size_t)N * 64 * 4);
  float*  sq  = (float*)alloc((size_t)N * 4);
  int*    nbr = (int*)  alloc((size_t)N * KNN * 4);
  __bf16* s2  = (__bf16*)alloc((size_t)N * 1216 * 2);
  __bf16* bb1 = (__bf16*)alloc((size_t)N * 256 * 2);
  __bf16* bb2 = (__bf16*)alloc((size_t)N * 256 * 2);
  __bf16* bb3 = (__bf16*)alloc((size_t)N * 128 * 2);
  __bf16* w1a_p = (__bf16*)alloc(32 * 64 * 2);
  __bf16* w1b_p = (__bf16*)alloc(64 * 64 * 2);
  __bf16* w2a_p = (__bf16*)alloc(128 * 64 * 2);
  __bf16* w2b_p = (__bf16*)alloc(64 * 64 * 2);
  __bf16* w3_p  = (__bf16*)alloc(128 * 64 * 2);
  __bf16* wf1_p = (__bf16*)alloc(192 * 1024 * 2);
  __bf16* wf2_p = (__bf16*)alloc(1216 * 256 * 2);
  __bf16* wf3_p = (__bf16*)alloc(256 * 256 * 2);
  __bf16* wf4_p = (__bf16*)alloc(256 * 128 * 2);

  auto pack = [&](const float* w, __bf16* pw, int Kd, int Nd) {
    pack_w_kernel<<<(Kd * Nd + 255) / 256, 256, 0, stream>>>(w, pw, Kd, Nd);
  };
  pack(w1a, w1a_p, 32, 64);    pack(w1b, w1b_p, 64, 64);
  pack(w2a, w2a_p, 128, 64);   pack(w2b, w2b_p, 64, 64);
  pack(w3,  w3_p,  128, 64);
  pack(wf1, wf1_p, 192, 1024); pack(wf2, wf2_p, 1216, 256);
  pack(wf3, wf3_p, 256, 256);  pack(wf4, wf4_p, 256, 128);

  // Init accumulators (atomicMax-as-uint needs 0-init; also matches where(isfinite,.,0)).
  zero_f32_kernel<<<(N * 64 + 255) / 256, 256, 0, stream>>>(x1, N * 64);
  zero_f32_kernel<<<(N * 64 + 255) / 256, 256, 0, stream>>>(yb, N * 64);
  zero_f32_kernel<<<(N * 64 + 255) / 256, 256, 0, stream>>>(zb, N * 64);
  zero_f32_kernel<<<1, 256, 0, stream>>>(out, 1);

  // Stage 1: edge MLP + segment max  (E/16 = 10000 tiles, 8 waves/block)
  edge_mlp1_kernel<<<E / 16 / 8, 256, 0, stream>>>(x, ei, w1a_p, b1a, w1b_p, b1b, x1, E);

  // Stage 2: KNN(x1) fused Gram+top-60, then neighbor MLP -> y
  sqnorm_kernel<<<(N + 255) / 256, 256, 0, stream>>>(x1, sq, N);
  knn_topk_kernel<<<N / 16, 32, 0, stream>>>(x1, sq, nbr, N);
  edge_mlp2_kernel<<<(N * KNN / 16) / 4, 128, 0, stream>>>(x1, nbr, w2a_p, b2a, w2b_p, b2b, yb);

  // Stage 3: KNN(y), single layer -> z
  sqnorm_kernel<<<(N + 255) / 256, 256, 0, stream>>>(yb, sq, N);
  knn_topk_kernel<<<N / 16, 32, 0, stream>>>(yb, sq, nbr, N);
  edge_mlp3_kernel<<<(N * KNN / 16) / 4, 128, 0, stream>>>(yb, nbr, w3_p, b3w, zb);

  // Dense head (all WMMA bf16): s1 -> a (written into s2 cols 192..1215) -> b chain
  concat_s1_kernel<<<(N * 192 + 255) / 256, 256, 0, stream>>>(x1, yb, zb, s2, N);
  {
    int waves;
    waves = (N / 16) * (1024 / 64);   // 192 -> 1024
    gemm_bias_act_kernel<<<(waves + 7) / 8, 256, 0, stream>>>(s2, 1216, wf1_p, bf1,
                                                              s2 + 192, 1216, N, 192, 1024, 1);
    waves = (N / 16) * (256 / 64);    // 1216 -> 256
    gemm_bias_act_kernel<<<(waves + 7) / 8, 256, 0, stream>>>(s2, 1216, wf2_p, bf2,
                                                              bb1, 256, N, 1216, 256, 1);
    waves = (N / 16) * (256 / 64);    // 256 -> 256
    gemm_bias_act_kernel<<<(waves + 7) / 8, 256, 0, stream>>>(bb1, 256, wf3_p, bf3,
                                                              bb2, 256, N, 256, 256, 1);
    waves = (N / 16) * (128 / 64);    // 256 -> 128
    gemm_bias_act_kernel<<<(waves + 7) / 8, 256, 0, stream>>>(bb2, 256, wf4_p, bf4,
                                                              bb3, 128, N, 256, 128, 1);
  }
  // 128 -> 1, sigmoid, BCE loss
  final_loss_kernel<<<(N + 255) / 256, 256, 0, stream>>>(bb3, wf5, bf5, labels, out, N);

  (void)in_sizes; (void)n_in; (void)out_size; (void)ws_size;
}
